// LSTMModel_90288802496776
// MI455X (gfx1250) — matrix operand order repaired and checked by reading, not compiled, not run on hardware
//
#include <hip/hip_runtime.h>

#define B_    256
#define T_    512
#define D_    75
#define H_    512
#define G4_   2048
#define NCLS_ 11
#define NBLK  128
#define NTHR  128
#define KP_HH 520   // padded LDS K-stride for w_hh slice (bank-conflict free)
#define KP_IH 104   // padded LDS K-stride for w_ih slice (K padded 75 -> 96)
#define XP_   96    // padded K for bf16 x copy

typedef __attribute__((ext_vector_type(16))) __bf16 bf16x16;
typedef __attribute__((ext_vector_type(8)))  float  f32x8;

union FragAB {
    bf16x16 v;
    uint4 u[2];
    unsigned short s[16];
};

__device__ __forceinline__ unsigned short f2bf(float f) {
    unsigned u = __float_as_uint(f);
    unsigned r = (u + 0x7FFFu + ((u >> 16) & 1u)) >> 16;  // round-to-nearest-even
    return (unsigned short)r;
}
__device__ __forceinline__ float softplusf(float x) {
    return (x > 20.0f) ? x : log1pf(__expf(x));
}

// gfx1250 hardware transcendental tanh (TRANS32 op).
__device__ __forceinline__ float tanh_fast(float x) {
#if defined(__has_builtin) && __has_builtin(__builtin_amdgcn_tanhf)
    return __builtin_amdgcn_tanhf(x);
#else
    float r;
    asm volatile("v_tanh_f32 %0, %1\n\tv_nop" : "=v"(r) : "v"(x));
    return r;
#endif
}
// sigmoid(x + b) = 0.5 + 0.5*tanh(0.5*x + 0.5*b): bias folds into the FMA.
__device__ __forceinline__ float sigmoid_biased(float x, float half_b) {
    return fmaf(0.5f, tanh_fast(fmaf(0.5f, x, half_b)), 0.5f);
}

__global__ void init_barrier(unsigned* bar) {
    if (threadIdx.x == 0) { bar[0] = 0u; bar[1] = 0u; }
}

// One-time fp32 -> bf16 convert of x with K padded 75 -> 96 (zero fill).
__global__ void xpad_kernel(const float* __restrict__ x, unsigned short* __restrict__ xpad) {
    int row = blockIdx.x;                 // b*T + t
    int d   = threadIdx.x;                // 0..95
    unsigned short v = 0;
    if (d < D_) v = f2bf(x[(size_t)row * D_ + d]);
    xpad[(size_t)row * XP_ + d] = v;
}

// Persistent LSTM: 128 blocks x 128 threads (4 waves). Block = (jt, bgrp):
// jt = hidden-column tile (16 cols), bgrp picks 64 batch rows; each wave owns
// a 16x16 (batch x hidden) tile and its 4 gate tiles. w_hh/w_ih slices live in
// LDS for all 512 steps; cell state stays in WMMA accumulator registers.
__global__ void lstm_persistent(
    const unsigned short* __restrict__ xpad,   // [B][T][XP_] bf16
    const float* __restrict__ wih_mu, const float* __restrict__ wih_rho, const float* __restrict__ eps_ih,
    const float* __restrict__ whh_mu, const float* __restrict__ whh_rho, const float* __restrict__ eps_hh,
    const float* __restrict__ b_mu,   const float* __restrict__ b_rho,   const float* __restrict__ eps_b,
    unsigned* __restrict__ bar,            // [0]=arrive count, [1]=generation
    unsigned short* __restrict__ hbuf0,    // [B][H] bf16 ping
    unsigned short* __restrict__ hbuf1,    // [B][H] bf16 pong
    float* __restrict__ hfinal)            // [B][H] f32
{
    extern __shared__ __align__(16) unsigned char smem_raw[];
    unsigned short* sWhh = (unsigned short*)smem_raw;   // [64][KP_HH] (col = g*16+jc, K-major)
    unsigned short* sWih = sWhh + 64 * KP_HH;           // [64][KP_IH]

    const int jt    = blockIdx.x >> 2;   // 0..31 hidden-col tile
    const int bgrp  = blockIdx.x & 3;    // 0..3  -> batch rows [bgrp*64, +64)
    const int tid   = threadIdx.x;
    const int lane  = tid & 31;
    const int wv    = tid >> 5;          // 0..3
    const int n     = lane & 15;         // fragment column / A fragment row
    const int hi    = lane >> 4;
    const int btile = bgrp * 4 + wv;     // 0..15
    const int rowA  = btile * 16 + n;    // global batch row for A fragment

    // ---- Phase 0: sample (mu + softplus(rho)*eps) -> bf16 LDS slices ----
    for (int i = tid; i < 64 * H_; i += NTHR) {
        int col = i >> 9;                 // 0..63
        int k   = i & 511;
        int g   = col >> 4, jc = col & 15;
        int gi  = k * G4_ + g * H_ + jt * 16 + jc;
        float v = whh_mu[gi] + softplusf(whh_rho[gi]) * eps_hh[gi];
        sWhh[col * KP_HH + k] = f2bf(v);
    }
    for (int i = tid; i < 64 * 96; i += NTHR) {
        int col = i / 96;
        int k   = i - col * 96;
        int g   = col >> 4, jc = col & 15;
        float v = 0.0f;
        if (k < D_) {
            int gi = k * G4_ + g * H_ + jt * 16 + jc;
            v = wih_mu[gi] + softplusf(wih_rho[gi]) * eps_ih[gi];
        }
        sWih[col * KP_IH + k] = f2bf(v);
    }
    __syncthreads();

    // Sampled bias for this lane's output column (i,f,g,o); pre-scale for the
    // tanh-based sigmoid so the bias-add fuses into an FMA in the epilogue.
    float bh0, bh1, b2, bh3;
    {
        int gc0 = 0 * H_ + jt * 16 + n;
        int gc1 = 1 * H_ + jt * 16 + n;
        int gc2 = 2 * H_ + jt * 16 + n;
        int gc3 = 3 * H_ + jt * 16 + n;
        bh0 = 0.5f * (b_mu[gc0] + softplusf(b_rho[gc0]) * eps_b[gc0]);
        bh1 = 0.5f * (b_mu[gc1] + softplusf(b_rho[gc1]) * eps_b[gc1]);
        b2  =        (b_mu[gc2] + softplusf(b_rho[gc2]) * eps_b[gc2]);
        bh3 = 0.5f * (b_mu[gc3] + softplusf(b_rho[gc3]) * eps_b[gc3]);
    }

    // Per-lane LDS fragment bases (column n, K-chunk by lane half).
    const unsigned short* sWihN = sWih + n * KP_IH + hi * 16;
    const unsigned short* sWhhN = sWhh + n * KP_HH + hi * 16;

    const unsigned short* xrow = xpad + (size_t)rowA * T_ * XP_;  // this lane's A rows

    f32x8 cacc = {};  // cell state tile, persistent in accumulator layout

    for (int t = 0; t < T_; ++t) {
        const unsigned short* hprev = (t & 1) ? hbuf1 : hbuf0;
        unsigned short*       hnext = (t & 1) ? hbuf0 : hbuf1;

        f32x8 acc0, acc1, acc2, acc3;

        // ---- x-projection: A = xpad[rowA, t, :] (bf16, pre-padded to 96) ----
        {
            const unsigned short* xr = xrow + (size_t)t * XP_;
            // kk = 0: first WMMA group produces the accumulators from zero C
            // (no live-in accumulators -> no spill pressure at the load cluster)
            {
                const f32x8 zc = {};
                FragAB a;
                a.u[0] = *(const uint4*)(xr + hi * 16);
                a.u[1] = *(const uint4*)(xr + hi * 16 + 8);
                FragAB b0, b1, b2f, b3;
                b0.u[0]  = *(const uint4*)(sWihN + 0 * 16 * KP_IH);
                b0.u[1]  = *(const uint4*)(sWihN + 0 * 16 * KP_IH + 8);
                b1.u[0]  = *(const uint4*)(sWihN + 1 * 16 * KP_IH);
                b1.u[1]  = *(const uint4*)(sWihN + 1 * 16 * KP_IH + 8);
                b2f.u[0] = *(const uint4*)(sWihN + 2 * 16 * KP_IH);
                b2f.u[1] = *(const uint4*)(sWihN + 2 * 16 * KP_IH + 8);
                b3.u[0]  = *(const uint4*)(sWihN + 3 * 16 * KP_IH);
                b3.u[1]  = *(const uint4*)(sWihN + 3 * 16 * KP_IH + 8);
                acc0 = __builtin_amdgcn_wmma_f32_16x16x32_bf16(false, a.v, false, b0.v,  (short)0, zc, false, false);
                acc1 = __builtin_amdgcn_wmma_f32_16x16x32_bf16(false, a.v, false, b1.v,  (short)0, zc, false, false);
                acc2 = __builtin_amdgcn_wmma_f32_16x16x32_bf16(false, a.v, false, b2f.v, (short)0, zc, false, false);
                acc3 = __builtin_amdgcn_wmma_f32_16x16x32_bf16(false, a.v, false, b3.v,  (short)0, zc, false, false);
            }
#pragma unroll
            for (int kk = 1; kk < 3; ++kk) {
                int kb = kk * 32 + hi * 16;
                FragAB a;
                a.u[0] = *(const uint4*)(xr + kb);
                a.u[1] = *(const uint4*)(xr + kb + 8);
                FragAB b0, b1, b2f, b3;
                b0.u[0]  = *(const uint4*)(sWihN + 0 * 16 * KP_IH + kk * 32);
                b0.u[1]  = *(const uint4*)(sWihN + 0 * 16 * KP_IH + kk * 32 + 8);
                b1.u[0]  = *(const uint4*)(sWihN + 1 * 16 * KP_IH + kk * 32);
                b1.u[1]  = *(const uint4*)(sWihN + 1 * 16 * KP_IH + kk * 32 + 8);
                b2f.u[0] = *(const uint4*)(sWihN + 2 * 16 * KP_IH + kk * 32);
                b2f.u[1] = *(const uint4*)(sWihN + 2 * 16 * KP_IH + kk * 32 + 8);
                b3.u[0]  = *(const uint4*)(sWihN + 3 * 16 * KP_IH + kk * 32);
                b3.u[1]  = *(const uint4*)(sWihN + 3 * 16 * KP_IH + kk * 32 + 8);
                acc0 = __builtin_amdgcn_wmma_f32_16x16x32_bf16(false, a.v, false, b0.v,  (short)0, acc0, false, false);
                acc1 = __builtin_amdgcn_wmma_f32_16x16x32_bf16(false, a.v, false, b1.v,  (short)0, acc1, false, false);
                acc2 = __builtin_amdgcn_wmma_f32_16x16x32_bf16(false, a.v, false, b2f.v, (short)0, acc2, false, false);
                acc3 = __builtin_amdgcn_wmma_f32_16x16x32_bf16(false, a.v, false, b3.v,  (short)0, acc3, false, false);
            }
        }

        // ---- recurrent h @ w_hh (h0 == 0, so skip at t==0) ----
        if (t > 0) {
            const unsigned short* hr = hprev + (size_t)rowA * H_;
#pragma unroll 4
            for (int kk = 0; kk < 16; ++kk) {
                int kb = kk * 32 + hi * 16;
                FragAB a;
                a.u[0] = *(const uint4*)(hr + kb);
                a.u[1] = *(const uint4*)(hr + kb + 8);
                FragAB b0, b1, b2f, b3;
                b0.u[0]  = *(const uint4*)(sWhhN + 0 * 16 * KP_HH + kk * 32);
                b0.u[1]  = *(const uint4*)(sWhhN + 0 * 16 * KP_HH + kk * 32 + 8);
                b1.u[0]  = *(const uint4*)(sWhhN + 1 * 16 * KP_HH + kk * 32);
                b1.u[1]  = *(const uint4*)(sWhhN + 1 * 16 * KP_HH + kk * 32 + 8);
                b2f.u[0] = *(const uint4*)(sWhhN + 2 * 16 * KP_HH + kk * 32);
                b2f.u[1] = *(const uint4*)(sWhhN + 2 * 16 * KP_HH + kk * 32 + 8);
                b3.u[0]  = *(const uint4*)(sWhhN + 3 * 16 * KP_HH + kk * 32);
                b3.u[1]  = *(const uint4*)(sWhhN + 3 * 16 * KP_HH + kk * 32 + 8);
                acc0 = __builtin_amdgcn_wmma_f32_16x16x32_bf16(false, a.v, false, b0.v,  (short)0, acc0, false, false);
                acc1 = __builtin_amdgcn_wmma_f32_16x16x32_bf16(false, a.v, false, b1.v,  (short)0, acc1, false, false);
                acc2 = __builtin_amdgcn_wmma_f32_16x16x32_bf16(false, a.v, false, b2f.v, (short)0, acc2, false, false);
                acc3 = __builtin_amdgcn_wmma_f32_16x16x32_bf16(false, a.v, false, b3.v,  (short)0, acc3, false, false);
            }
        }

        // ---- gates -> c, h via hardware tanh; bias folded into FMAs ----
#pragma unroll
        for (int e = 0; e < 8; ++e) {
            float ig = sigmoid_biased(acc0[e], bh0);
            float fg = sigmoid_biased(acc1[e], bh1);
            float gg = tanh_fast(acc2[e] + b2);
            float og = sigmoid_biased(acc3[e], bh3);
            float c  = fmaf(fg, cacc[e], ig * gg);
            cacc[e]  = c;
            float h  = og * tanh_fast(c);
            int brow = btile * 16 + e + hi * 8;
            int jcol = jt * 16 + n;
            hnext[(size_t)brow * H_ + jcol] = f2bf(h);
            if (t == T_ - 1) hfinal[(size_t)brow * H_ + jcol] = h;
        }

        // ---- warm next step's x row while we wait (independent of barrier) ----
        if (t < T_ - 1) {
            __builtin_prefetch(xrow + (size_t)(t + 1) * XP_, 0, 0);

            // ---- device-wide sense barrier between steps ----
            __syncthreads();
            if (tid == 0) {
                __threadfence();
                unsigned arr = atomicAdd(&bar[0], 1u);
                if (arr == NBLK - 1) {
                    atomicExch(&bar[0], 0u);
                    __threadfence();
                    atomicAdd(&bar[1], 1u);
                } else {
                    while (__atomic_load_n(&bar[1], __ATOMIC_ACQUIRE) < (unsigned)(t + 1)) {
                        __builtin_amdgcn_s_sleep(1);
                    }
                }
                __threadfence();
            }
            __syncthreads();
        }
    }
}

// Tiny deterministic final FC: z[b,c] = h[b,:] . fc_w[c,:] + fc_b[c]
__global__ void fc_kernel(const float* __restrict__ hfin,
                          const float* __restrict__ fc_w,
                          const float* __restrict__ fc_b,
                          float* __restrict__ out) {
    int b = blockIdx.x;
    int c = threadIdx.x;
    if (c < NCLS_) {
        const float* hr = hfin + (size_t)b * H_;
        const float* wr = fc_w + (size_t)c * H_;
        float acc = fc_b[c];
        for (int j = 0; j < H_; ++j) acc = fmaf(hr[j], wr[j], acc);
        out[b * NCLS_ + c] = acc;
    }
}

extern "C" void kernel_launch(void* const* d_in, const int* in_sizes, int n_in,
                              void* d_out, int out_size, void* d_ws, size_t ws_size,
                              hipStream_t stream) {
    const float* x       = (const float*)d_in[0];
    const float* wih_mu  = (const float*)d_in[1];
    const float* wih_rho = (const float*)d_in[2];
    const float* eps_ih  = (const float*)d_in[3];
    const float* whh_mu  = (const float*)d_in[4];
    const float* whh_rho = (const float*)d_in[5];
    const float* eps_hh  = (const float*)d_in[6];
    const float* b_mu    = (const float*)d_in[7];
    const float* b_rho   = (const float*)d_in[8];
    const float* eps_b   = (const float*)d_in[9];
    const float* fc_w    = (const float*)d_in[10];
    const float* fc_b    = (const float*)d_in[11];

    unsigned char* ws = (unsigned char*)d_ws;
    size_t off = 0;
    unsigned*       bar    = (unsigned*)(ws + off);        off += 1024;
    unsigned short* hbuf0  = (unsigned short*)(ws + off);  off += (size_t)B_ * H_ * 2;
    unsigned short* hbuf1  = (unsigned short*)(ws + off);  off += (size_t)B_ * H_ * 2;
    float*          hfinal = (float*)(ws + off);           off += (size_t)B_ * H_ * 4;
    unsigned short* xpad   = (unsigned short*)(ws + off);  off += (size_t)B_ * T_ * XP_ * 2;

    init_barrier<<<1, 32, 0, stream>>>(bar);
    xpad_kernel<<<dim3(B_ * T_), dim3(XP_), 0, stream>>>(x, xpad);

    size_t shmem = (size_t)(64 * KP_HH + 64 * KP_IH) * sizeof(unsigned short); // ~78 KB
    lstm_persistent<<<dim3(NBLK), dim3(NTHR), shmem, stream>>>(
        xpad, wih_mu, wih_rho, eps_ih, whh_mu, whh_rho, eps_hh,
        b_mu, b_rho, eps_b, bar, hbuf0, hbuf1, hfinal);

    fc_kernel<<<dim3(B_), dim3(32), 0, stream>>>(hfinal, fc_w, fc_b, (float*)d_out);
}
